// TextEncoder_69784628626092
// MI455X (gfx1250) — matrix operand order repaired and checked by reading, not compiled, not run on hardware
//
#include <hip/hip_runtime.h>
#include <cstdint>
#include <cstddef>

// ---- problem constants (match reference) ----
#define NLAYER 4
#define CC     512
#define HH     8
#define DD     64
#define FCC    2048
#define WWIN   4
#define BB     4
#define LL     1024
#define NR     (2*WWIN + 1)     // 9 relative positions
#define QSCALE 0.125f           // 64^-0.5
#define EPSV   1e-6f

typedef __attribute__((ext_vector_type(16))) __bf16 v16bf;
typedef __attribute__((ext_vector_type(8)))  __bf16 v8bf;
typedef __attribute__((ext_vector_type(8)))  float  v8f;
typedef __attribute__((ext_vector_type(4)))  float  v4f;

__device__ __forceinline__ v16bf cat8(v8bf lo, v8bf hi) {
    return __builtin_shufflevector(lo, hi, 0,1,2,3,4,5,6,7,8,9,10,11,12,13,14,15);
}

__device__ __forceinline__ v8f wmma_bf16(v16bf a, v16bf b, v8f c) {
    // D = A(16x32 bf16) * B(32x16 bf16) + C(16x16 f32)
    return __builtin_amdgcn_wmma_f32_16x16x32_bf16(false, a, false, b, (short)0, c, false, false);
}

// CDNA5 async global->LDS copy (ASYNCcnt path), 16B per lane.
__device__ __forceinline__ void async_copy_b128(uint32_t lds_off, const void* gaddr) {
    asm volatile("global_load_async_to_lds_b128 %0, %1, off"
                 :: "v"(lds_off), "v"(gaddr) : "memory");
}
__device__ __forceinline__ void wait_async0() {
    asm volatile("s_wait_asynccnt 0x0" ::: "memory");
}

// ============================================================
// elementwise: dst = src * mask (mask is [B,1,L])
// ============================================================
__global__ void mask_mul_kernel(float* __restrict__ dst, const float* __restrict__ src,
                                const float* __restrict__ mask) {
    int i = blockIdx.x * 256 + threadIdx.x;         // over B*C*L
    int l = i % LL;
    int b = i / (CC * LL);
    dst[i] = src[i] * mask[b * LL + l];
}

// ============================================================
// one-shot f32 -> bf16 weight conversion
// ============================================================
__global__ void conv_bf16_kernel(__bf16* __restrict__ dst, const float* __restrict__ src, int n) {
    int i = blockIdx.x * 256 + threadIdx.x;
    if (i < n) dst[i] = (__bf16)src[i];
}

// ============================================================
// WMMA GEMM: out[b][o][l] = act( W[o][:] . X[b][:][l] + bias[o] )
// W: [Cout][Cin] bf16 (pre-converted) -- staged to LDS via
//    global_load_async_to_lds_b128 (ASYNCcnt), DOUBLE BUFFERED so
//    the async copy + B-tile loads overlap the WMMAs of the
//    previous K-step (one barrier per K-step).
// X: [B][Cin][L] f32, staged f32->bf16 transposed into LDS.
// Block: 256 thr (8 waves). Block tile 64(M) x 128(N), K-step 32.
// Wave tile: 32 x 32 -> four 16x16 WMMA accumulators, 4 WMMA/K-step.
// EPI: 0 = f32 +bias
//      1 = f32 relu(+bias)*mask
//      2 = f32 (+bias)*mask
//      3 = bf16 q: (v+bias)*QSCALE  -> [B][H][L][D]
//      4 = bf16 k:  v+bias          -> [B][H][L][D]
//      5 = bf16 v:  v+bias          -> [B][C][L]
// ============================================================
template<int EPI, bool MASK_IN>
__global__ __launch_bounds__(256)
void gemm_kernel(const __bf16* __restrict__ Wb, const float* __restrict__ bias,
                 const float* __restrict__ X, const float* __restrict__ mask,
                 void* __restrict__ outp, int Cin, int Cout) {
    const int b  = blockIdx.z;
    const int m0 = blockIdx.y * 64;      // output-channel tile
    const int l0 = blockIdx.x * 128;     // sequence tile

    const int tid  = threadIdx.x;
    const int lane = tid & 31;
    const int wave = tid >> 5;
    const int wm   = (wave & 1) * 32;    // wave row offset in block tile
    const int wn   = (wave >> 1) * 32;   // wave col offset in block tile
    const int lane15 = lane & 15;
    const int half   = lane >> 4;

    __shared__ __bf16 As[2][64][40];     // [buf][m][k], pad to 40 for aligned 16B chunks
    __shared__ __bf16 Bt[2][128][40];    // [buf][n][k] transposed activations

    v8f acc[2][2];
    #pragma unroll
    for (int mi = 0; mi < 2; ++mi)
        #pragma unroll
        for (int ni = 0; ni < 2; ++ni) acc[mi][ni] = v8f{};

    // staging assignments (fixed per thread)
    const int arow = tid >> 2;                    // 0..63
    const int akc  = (tid & 3) * 8;               // 0,8,16,24 (elements)
    const uint32_t lds_a[2] = { (uint32_t)(uintptr_t)&As[0][arow][akc],
                                (uint32_t)(uintptr_t)&As[1][arow][akc] };
    const int bkk  = tid >> 3;                    // 0..31
    const int bnc  = (tid & 7) * 16;              // 0..112

    // mask values for B staging are kk-invariant: hoist out of the K-loop
    v4f mk[4];
    if (MASK_IN) {
        const float* mp = mask + (size_t)b * LL + l0 + bnc;
        mk[0] = *(const v4f*)(mp);
        mk[1] = *(const v4f*)(mp + 4);
        mk[2] = *(const v4f*)(mp + 8);
        mk[3] = *(const v4f*)(mp + 12);
    }

    auto stage = [&](int kk, int buf) {
        // A: bf16 weights via async copy (raw bytes, no conversion)
        async_copy_b128(lds_a[buf], Wb + (size_t)(m0 + arow) * Cin + kk + akc);
        // B: activations, f32 -> bf16, transposed
        const float* src = X + ((size_t)b * Cin + kk + bkk) * LL + l0 + bnc;
        v4f x[4];
        x[0] = *(const v4f*)(src);
        x[1] = *(const v4f*)(src + 4);
        x[2] = *(const v4f*)(src + 8);
        x[3] = *(const v4f*)(src + 12);
        #pragma unroll
        for (int i = 0; i < 16; ++i) {
            float xv = x[i >> 2][i & 3];
            if (MASK_IN) xv *= mk[i >> 2][i & 3];
            Bt[buf][bnc + i][bkk] = (__bf16)xv;
        }
    };

    stage(0, 0);
    int cur = 0;
    for (int kk = 0; kk < Cin; kk += 32) {
        wait_async0();                     // own async copies for buf[cur] done
        __syncthreads();                   // all waves' staging of buf[cur] visible
        if (kk + 32 < Cin) stage(kk + 32, cur ^ 1);   // overlap with compute below

        // ---- fragments + 4 WMMAs from buf[cur] ----
        const int kb = half * 8;           // A: K chunks {kb..kb+7, kb+16..kb+23}
        const int ks = half * 8;           // B: K chunks {ks..ks+7, ks+16..ks+23}
        v16bf af[2], bfr[2];
        #pragma unroll
        for (int s = 0; s < 2; ++s) {
            af[s]  = cat8(*(const v8bf*)&As[cur][wm + s*16 + lane15][kb],
                          *(const v8bf*)&As[cur][wm + s*16 + lane15][kb + 16]);
            bfr[s] = cat8(*(const v8bf*)&Bt[cur][wn + s*16 + lane15][ks],
                          *(const v8bf*)&Bt[cur][wn + s*16 + lane15][ks + 16]);
        }
        #pragma unroll
        for (int mi = 0; mi < 2; ++mi)
            #pragma unroll
            for (int ni = 0; ni < 2; ++ni)
                acc[mi][ni] = wmma_bf16(af[mi], bfr[ni], acc[mi][ni]);
        cur ^= 1;
    }

    // ---- epilogue ----
    #pragma unroll
    for (int mi = 0; mi < 2; ++mi) {
        #pragma unroll
        for (int ni = 0; ni < 2; ++ni) {
            int l = l0 + wn + ni * 16 + lane15;
            float mv = (EPI == 1 || EPI == 2) ? mask[(size_t)b * LL + l] : 1.0f;
            #pragma unroll
            for (int r = 0; r < 8; ++r) {
                int o = m0 + wm + mi * 16 + half * 8 + r;
                float v = acc[mi][ni][r] + bias[o];
                if (EPI == 0) {
                    ((float*)outp)[((size_t)b * Cout + o) * LL + l] = v;
                } else if (EPI == 1) {
                    v = v > 0.0f ? v : 0.0f;
                    ((float*)outp)[((size_t)b * Cout + o) * LL + l] = v * mv;
                } else if (EPI == 2) {
                    ((float*)outp)[((size_t)b * Cout + o) * LL + l] = v * mv;
                } else if (EPI == 3) {
                    int h = o >> 6, d = o & 63;
                    ((__bf16*)outp)[(((size_t)b * HH + h) * LL + l) * DD + d] = (__bf16)(v * QSCALE);
                } else if (EPI == 4) {
                    int h = o >> 6, d = o & 63;
                    ((__bf16*)outp)[(((size_t)b * HH + h) * LL + l) * DD + d] = (__bf16)v;
                } else {
                    ((__bf16*)outp)[((size_t)b * CC + o) * LL + l] = (__bf16)v;
                }
            }
        }
    }
}

// ============================================================
// LayerNorm over channels of (a + res), write out. out may alias res.
// ============================================================
__global__ __launch_bounds__(128)
void ln_add_kernel(const float* __restrict__ a, const float* __restrict__ res,
                   const float* __restrict__ g, const float* __restrict__ beta,
                   float* __restrict__ outp) {
    int col = blockIdx.x;                 // B*L
    int b = col / LL, l = col % LL;
    int tid = threadIdx.x;
    float vals[4];
    float s = 0.f, ss = 0.f;
    #pragma unroll
    for (int i = 0; i < 4; ++i) {
        int c = tid + i * 128;
        size_t idx = ((size_t)b * CC + c) * LL + l;
        float v = a[idx] + res[idx];
        vals[i] = v; s += v; ss += v * v;
    }
    #pragma unroll
    for (int off = 16; off > 0; off >>= 1) {
        s  += __shfl_xor(s,  off, 32);
        ss += __shfl_xor(ss, off, 32);
    }
    __shared__ float sh[10];
    int wv = tid >> 5, ln = tid & 31;
    if (ln == 0) { sh[wv] = s; sh[4 + wv] = ss; }
    __syncthreads();
    if (tid == 0) {
        float ts = sh[0] + sh[1] + sh[2] + sh[3];
        float tss = sh[4] + sh[5] + sh[6] + sh[7];
        float mean = ts * (1.0f / CC);
        float var  = tss * (1.0f / CC) - mean * mean;
        sh[8] = mean;
        sh[9] = rsqrtf(var + EPSV);
    }
    __syncthreads();
    float mean = sh[8], rs = sh[9];
    #pragma unroll
    for (int i = 0; i < 4; ++i) {
        int c = tid + i * 128;
        outp[((size_t)b * CC + c) * LL + l] = (vals[i] - mean) * rs * g[c] + beta[c];
    }
}

// ============================================================
// qrk[b,h,l,r] = q[b,h,l,:] . rel_k[r,:]   (q already scaled)
// ============================================================
__global__ void qrk_kernel(const __bf16* __restrict__ qb, const float* __restrict__ relk,
                           float* __restrict__ qrk) {
    int id = blockIdx.x * 256 + threadIdx.x;
    if (id >= BB * HH * LL * NR) return;
    int r = id % NR;
    int l = (id / NR) % LL;
    int h = (id / (NR * LL)) % HH;
    int b = id / (NR * LL * HH);
    const __bf16* q  = qb + (((size_t)b * HH + h) * LL + l) * DD;
    const float*  rk = relk + (size_t)r * DD;
    float acc = 0.f;
    #pragma unroll 8
    for (int d = 0; d < DD; ++d) acc += (float)q[d] * rk[d];
    qrk[id] = acc;
}

// ============================================================
// Flash attention: one wave per (b, h, 16-query tile).
// q,k in [B][H][L][D] bf16 ; v in [B][C][L] bf16 (c = h*D + d).
// Streams 32-key blocks: S = QK^T (4 WMMA), online softmax,
// P (via LDS bf16) x V (4 WMMA). rel_k bias added only for the
// <=2 key blocks that overlap the +/-4 window (wave-uniform test,
// branchless per-element select). Saves softmax stats.
// ============================================================
__global__ __launch_bounds__(32)
void attn_kernel(const __bf16* __restrict__ qb, const __bf16* __restrict__ kbuf,
                 const __bf16* __restrict__ vb, const float* __restrict__ qrkb,
                 const float* __restrict__ mask, float* __restrict__ attnout,
                 float* __restrict__ mstat, float* __restrict__ lstat) {
    const int blk = blockIdx.x;                     // B*H*(L/16)
    const int qt = blk % (LL / 16);
    const int h  = (blk / (LL / 16)) % HH;
    const int b  = blk / ((LL / 16) * HH);
    const int lane = threadIdx.x;
    const int lane15 = lane & 15;
    const int half = lane >> 4;
    const int l0 = qt * 16;

    const __bf16* qbase = qb   + ((size_t)b * HH + h) * LL * DD;
    const __bf16* kbase = kbuf + ((size_t)b * HH + h) * LL * DD;
    const __bf16* vbase = vb   + ((size_t)b * CC + h * DD) * LL;
    const float*  qrkrow = qrkb + ((size_t)b * HH + h) * LL * NR;

    // Q A-fragments for K-dim chunks d=[0,32) and d=[32,64)
    const int kb0 = half ? 8 : 0;
    v16bf aq[2];
    #pragma unroll
    for (int t = 0; t < 2; ++t) {
        const __bf16* p = qbase + (size_t)(l0 + lane15) * DD + t * 32;
        aq[t] = cat8(*(const v8bf*)(p + kb0), *(const v8bf*)(p + kb0 + 16));
    }

    v8f O[4];
    #pragma unroll
    for (int j = 0; j < 4; ++j) O[j] = v8f{};
    float mi[8], li[8];
    #pragma unroll
    for (int r = 0; r < 8; ++r) { mi[r] = -3.0e38f; li[r] = 0.f; }

    __shared__ __bf16 Pt[16][40];

    for (int m0 = 0; m0 < LL; m0 += 32) {
        // wave-uniform: does this key block touch the +/-W window of rows l0..l0+15?
        const bool nearq = (m0 + 31 >= l0 - WWIN) && (m0 <= l0 + 15 + WWIN);

        // ---- S = Q K^T for two 16-key halves ----
        v8f S[2];
        #pragma unroll
        for (int t = 0; t < 2; ++t) {
            int m = m0 + t * 16 + lane15;
            int ks = half ? 8 : 0;
            const __bf16* p0 = kbase + (size_t)m * DD + ks;        // d = ks..ks+7, ks+16..ks+23
            const __bf16* p1 = kbase + (size_t)m * DD + 32 + ks;   // d = 32+ks.. (same pattern)
            v16bf bk0 = cat8(*(const v8bf*)p0, *(const v8bf*)(p0 + 16));
            v16bf bk1 = cat8(*(const v8bf*)p1, *(const v8bf*)(p1 + 16));
            v8f s = v8f{};
            s = wmma_bf16(aq[0], bk0, s);
            s = wmma_bf16(aq[1], bk1, s);
            float mskv = mask[(size_t)b * LL + m];
            if (nearq) {
                #pragma unroll
                for (int r = 0; r < 8; ++r) {
                    int l = l0 + half * 8 + r;
                    int diff = m - l;
                    int idx = diff + WWIN;
                    idx = idx < 0 ? 0 : (idx > 2*WWIN ? 2*WWIN : idx);   // clamp -> always-valid load
                    float qv = qrkrow[(size_t)l * NR + idx];
                    bool inwin = (diff >= -WWIN) && (diff <= WWIN);
                    s[r] += inwin ? qv : 0.0f;
                }
            }
            #pragma unroll
            for (int r = 0; r < 8; ++r) s[r] = (mskv == 0.0f) ? -1e4f : s[r];
            S[t] = s;
        }
        // ---- online softmax (row reductions across 16-lane groups) ----
        float fac[8];
        #pragma unroll
        for (int r = 0; r < 8; ++r) {
            float mx = fmaxf(S[0][r], S[1][r]);
            #pragma unroll
            for (int off = 8; off >= 1; off >>= 1) mx = fmaxf(mx, __shfl_xor(mx, off, 16));
            float nm = fmaxf(mi[r], mx);
            float f  = __expf(mi[r] - nm);
            float p0 = __expf(S[0][r] - nm);
            float p1 = __expf(S[1][r] - nm);
            S[0][r] = p0; S[1][r] = p1;
            float rs = p0 + p1;
            #pragma unroll
            for (int off = 8; off >= 1; off >>= 1) rs += __shfl_xor(rs, off, 16);
            li[r] = li[r] * f + rs;
            mi[r] = nm;
            fac[r] = f;
        }
        #pragma unroll
        for (int j = 0; j < 4; ++j)
            #pragma unroll
            for (int r = 0; r < 8; ++r) O[j][r] *= fac[r];

        // ---- reshape P through LDS: C-layout -> A-layout ----
        __syncthreads();
        #pragma unroll
        for (int t = 0; t < 2; ++t)
            #pragma unroll
            for (int r = 0; r < 8; ++r)
                Pt[half * 8 + r][t * 16 + lane15] = (__bf16)S[t][r];
        __syncthreads();
        v16bf pf = cat8(*(const v8bf*)&Pt[lane15][kb0],
                        *(const v8bf*)&Pt[lane15][kb0 + 16]);

        // ---- O += P V  (4 d-tiles of 16) ----
        #pragma unroll
        for (int j = 0; j < 4; ++j) {
            int d = j * 16 + lane15;
            const __bf16* p = vbase + (size_t)d * LL + m0 + half * 8;
            v16bf vf = cat8(*(const v8bf*)p, *(const v8bf*)(p + 16));
            O[j] = wmma_bf16(pf, vf, O[j]);
        }
    }

    // ---- finalize: O /= l ; store [B][C][L] f32 + softmax stats ----
    float inv[8];
    #pragma unroll
    for (int r = 0; r < 8; ++r) inv[r] = (li[r] > 0.f) ? (1.0f / li[r]) : 0.f;
    float* obase = attnout + ((size_t)b * CC + h * DD) * LL;
    #pragma unroll
    for (int j = 0; j < 4; ++j) {
        int d = j * 16 + lane15;
        #pragma unroll
        for (int r = 0; r < 8; ++r) {
            int l = l0 + half * 8 + r;
            obase[(size_t)d * LL + l] = O[j][r] * inv[r];
        }
    }
    if (lane15 == 0) {
        #pragma unroll
        for (int r = 0; r < 8; ++r) {
            int l = l0 + half * 8 + r;
            mstat[((size_t)b * HH + h) * LL + l] = mi[r];
            lstat[((size_t)b * HH + h) * LL + l] = li[r];
        }
    }
}

// ============================================================
// rel_v term: attnout[b, h*D+d, l] += sum_r p(l, l+r-W) * rel_v[r, d]
// ============================================================
__global__ __launch_bounds__(64)
void relv_kernel(const __bf16* __restrict__ qb, const __bf16* __restrict__ kbuf,
                 const float* __restrict__ qrkb, const float* __restrict__ relv,
                 const float* __restrict__ mask, const float* __restrict__ mstat,
                 const float* __restrict__ lstat, float* __restrict__ attnout) {
    int blk = blockIdx.x;                 // B*H*L
    int l = blk % LL;
    int h = (blk / LL) % HH;
    int b = blk / (LL * HH);
    int tid = threadIdx.x;
    __shared__ float p9[NR];
    if (tid < NR) {
        int m = l + tid - WWIN;
        float p = 0.f;
        if (m >= 0 && m < LL && mask[(size_t)b * LL + m] != 0.f) {
            const __bf16* q = qb   + (((size_t)b * HH + h) * LL + l) * DD;
            const __bf16* k = kbuf + (((size_t)b * HH + h) * LL + m) * DD;
            float s = 0.f;
            #pragma unroll 8
            for (int d = 0; d < DD; ++d) s += (float)q[d] * (float)k[d];
            s += qrkb[(((size_t)b * HH + h) * LL + l) * NR + tid];
            float mv = mstat[((size_t)b * HH + h) * LL + l];
            float lv = lstat[((size_t)b * HH + h) * LL + l];
            p = (lv > 0.f) ? __expf(s - mv) / lv : 0.f;
        }
        p9[tid] = p;
    }
    __syncthreads();
    int d = tid;                           // 0..63
    float acc = 0.f;
    #pragma unroll
    for (int r = 0; r < NR; ++r) acc += p9[r] * relv[(size_t)r * DD + d];
    attnout[((size_t)b * CC + h * DD + d) * LL + l] += acc;
}

// ============================================================
// host launch
// ============================================================
extern "C" void kernel_launch(void* const* d_in, const int* in_sizes, int n_in,
                              void* d_out, int out_size, void* d_ws, size_t ws_size,
                              hipStream_t stream) {
    const float* x     = (const float*)d_in[0];
    const float* mask  = (const float*)d_in[1];
    const float* wq    = (const float*)d_in[2];
    const float* bq    = (const float*)d_in[3];
    const float* wk    = (const float*)d_in[4];
    const float* bk    = (const float*)d_in[5];
    const float* wv    = (const float*)d_in[6];
    const float* bv    = (const float*)d_in[7];
    const float* wo    = (const float*)d_in[8];
    const float* bo    = (const float*)d_in[9];
    const float* rel_k = (const float*)d_in[10];
    const float* rel_v = (const float*)d_in[11];
    const float* ln1g  = (const float*)d_in[12];
    const float* ln1b  = (const float*)d_in[13];
    const float* w1    = (const float*)d_in[14];
    const float* b1    = (const float*)d_in[15];
    const float* w2    = (const float*)d_in[16];
    const float* b2    = (const float*)d_in[17];
    const float* ln2g  = (const float*)d_in[18];
    const float* ln2b  = (const float*)d_in[19];
    float* out = (float*)d_out;

    char* ws = (char*)d_ws;
    size_t off = 0;
    auto carve = [&](size_t bytes) -> void* {
        void* p = (void*)(ws + off);
        off += (bytes + 255) & ~(size_t)255;
        return p;
    };
    const size_t BCL = (size_t)BB * CC * LL;
    const size_t WCC = (size_t)NLAYER * CC * CC;    // per qkv/o weight tensor
    const size_t WFC = (size_t)NLAYER * FCC * CC;   // per ffn weight tensor
    float*  xbuf    = (float*)carve(BCL * sizeof(float));
    float*  tmp     = (float*)carve(BCL * sizeof(float));
    float*  hbuf    = (float*)carve((size_t)BB * FCC * LL * sizeof(float));
    float*  attnbuf = (float*)carve(BCL * sizeof(float));
    __bf16* qbw     = (__bf16*)carve(BCL * sizeof(__bf16));
    __bf16* kbw     = (__bf16*)carve(BCL * sizeof(__bf16));
    __bf16* vbw     = (__bf16*)carve(BCL * sizeof(__bf16));
    float*  qrkbuf  = (float*)carve((size_t)BB * HH * LL * NR * sizeof(float));
    float*  mstat   = (float*)carve((size_t)BB * HH * LL * sizeof(float));
    float*  lstat   = (float*)carve((size_t)BB * HH * LL * sizeof(float));
    __bf16* wqb     = (__bf16*)carve(WCC * sizeof(__bf16));
    __bf16* wkb     = (__bf16*)carve(WCC * sizeof(__bf16));
    __bf16* wvb     = (__bf16*)carve(WCC * sizeof(__bf16));
    __bf16* wob     = (__bf16*)carve(WCC * sizeof(__bf16));
    __bf16* w1b     = (__bf16*)carve(WFC * sizeof(__bf16));
    __bf16* w2b     = (__bf16*)carve(WFC * sizeof(__bf16));
    (void)in_sizes; (void)n_in; (void)out_size; (void)ws_size;

    // one-shot bf16 weight conversion
    conv_bf16_kernel<<<(int)((WCC + 255) / 256), 256, 0, stream>>>(wqb, wq, (int)WCC);
    conv_bf16_kernel<<<(int)((WCC + 255) / 256), 256, 0, stream>>>(wkb, wk, (int)WCC);
    conv_bf16_kernel<<<(int)((WCC + 255) / 256), 256, 0, stream>>>(wvb, wv, (int)WCC);
    conv_bf16_kernel<<<(int)((WCC + 255) / 256), 256, 0, stream>>>(wob, wo, (int)WCC);
    conv_bf16_kernel<<<(int)((WFC + 255) / 256), 256, 0, stream>>>(w1b, w1, (int)WFC);
    conv_bf16_kernel<<<(int)((WFC + 255) / 256), 256, 0, stream>>>(w2b, w2, (int)WFC);

    const int EW = BB * CC * LL / 256;
    mask_mul_kernel<<<EW, 256, 0, stream>>>(xbuf, x, mask);

    dim3 gC(LL / 128, CC / 64, BB);        // 512-out GEMMs
    dim3 gF(LL / 128, FCC / 64, BB);       // 2048-out GEMM

    for (int i = 0; i < NLAYER; ++i) {
        const __bf16* Wq = wqb + (size_t)i * CC * CC;  const float* Bq = bq + (size_t)i * CC;
        const __bf16* Wk = wkb + (size_t)i * CC * CC;  const float* Bk = bk + (size_t)i * CC;
        const __bf16* Wv = wvb + (size_t)i * CC * CC;  const float* Bv = bv + (size_t)i * CC;
        const __bf16* Wo = wob + (size_t)i * CC * CC;  const float* Bo = bo + (size_t)i * CC;
        const float* RK = rel_k + (size_t)i * NR * DD;
        const float* RV = rel_v + (size_t)i * NR * DD;
        const __bf16* W1 = w1b + (size_t)i * FCC * CC; const float* B1 = b1 + (size_t)i * FCC;
        const __bf16* W2 = w2b + (size_t)i * CC * FCC; const float* B2 = b2 + (size_t)i * CC;

        // q/k/v projections (bf16 outputs, q pre-scaled, attention-friendly layouts)
        gemm_kernel<3, false><<<gC, 256, 0, stream>>>(Wq, Bq, xbuf, mask, qbw, CC, CC);
        gemm_kernel<4, false><<<gC, 256, 0, stream>>>(Wk, Bk, xbuf, mask, kbw, CC, CC);
        gemm_kernel<5, false><<<gC, 256, 0, stream>>>(Wv, Bv, xbuf, mask, vbw, CC, CC);

        // windowed rel-k bias table
        qrk_kernel<<<(BB * HH * LL * NR + 255) / 256, 256, 0, stream>>>(qbw, RK, qrkbuf);

        // flash attention + rel-v correction
        attn_kernel<<<BB * HH * (LL / 16), 32, 0, stream>>>(qbw, kbw, vbw, qrkbuf, mask,
                                                            attnbuf, mstat, lstat);
        relv_kernel<<<BB * HH * LL, 64, 0, stream>>>(qbw, kbw, qrkbuf, RV, mask,
                                                     mstat, lstat, attnbuf);

        // output projection, residual + LN1
        gemm_kernel<0, false><<<gC, 256, 0, stream>>>(Wo, Bo, attnbuf, mask, tmp, CC, CC);
        ln_add_kernel<<<BB * LL, 128, 0, stream>>>(tmp, xbuf, ln1g + (size_t)i * CC,
                                                   ln1b + (size_t)i * CC, xbuf);

        // FFN (mask-in + relu*mask, then *mask), residual + LN2
        gemm_kernel<1, true><<<gF, 256, 0, stream>>>(W1, B1, xbuf, mask, hbuf, CC, FCC);
        gemm_kernel<2, false><<<gC, 256, 0, stream>>>(W2, B2, hbuf, mask, tmp, FCC, CC);
        ln_add_kernel<<<BB * LL, 128, 0, stream>>>(tmp, xbuf, ln2g + (size_t)i * CC,
                                                   ln2b + (size_t)i * CC, xbuf);
    }

    mask_mul_kernel<<<EW, 256, 0, stream>>>(out, xbuf, mask);
}
